// QANet_33054068310071
// MI455X (gfx1250) — matrix operand order repaired and checked by reading, not compiled, not run on hardware
//
#include <hip/hip_runtime.h>
#include <hip/hip_bf16.h>

typedef _Float16 v16h __attribute__((ext_vector_type(16)));
typedef float    v8f  __attribute__((ext_vector_type(8)));

#define NEGBIG (-1e30f)

// ---------------- WMMA fragment helpers (16x16x32 f16, wave32) ----------------
// A-frag (16M x 32K): lane l -> row M = l&15 ; VGPR v holds K pair
//   kbase(l,v) = (v<4 ? 2v : 8+2v) + 8*(l>>4)
// B-frag (32K x 16N): lane l -> col N = l&15 ; elements hold K in the same kbase order as A
static __device__ __forceinline__ int qa_kbase(int lane, int v) {
  return ((v < 4) ? (2 * v) : (8 + 2 * v)) + ((lane >> 4) << 3);
}

static __device__ __forceinline__ v16h qa_frag_a_lds(const _Float16* tile, int stride, int lane) {
  const _Float16* row = tile + (long)(lane & 15) * stride;
  v16h f;
#pragma unroll
  for (int v = 0; v < 8; ++v) {
    int kb = qa_kbase(lane, v);
    f[2 * v]     = row[kb];
    f[2 * v + 1] = row[kb + 1];
  }
  return f;
}

// contiguous 16-half fragment read (vectorizes to 2x ds_load_b128)
static __device__ __forceinline__ v16h qa_frag_contig(const _Float16* p) {
  v16h f;
#pragma unroll
  for (int j = 0; j < 16; ++j) f[j] = p[j];
  return f;
}

// async global->LDS copy of 16 bytes (CDNA5 path, tracked by ASYNCcnt)
static __device__ __forceinline__ void qa_async_b128(unsigned lds_off, const void* src) {
  asm volatile("global_load_async_to_lds_b128 %0, %1, off"
               :: "v"(lds_off), "v"((unsigned long long)src) : "memory");
}
static __device__ __forceinline__ void qa_wait_async() {
  asm volatile("s_wait_asynccnt 0" ::: "memory");
}
static __device__ __forceinline__ unsigned qa_lds_off(const void* p) {
  return (unsigned)(unsigned long long)p;  // low 32 bits of LDS aperture = LDS offset
}

// ---------------- Generic fused GEMM: C = act(A*B + bias) + res ----------------
// A: (M,K) row-major (or (K,M) if transA); B: (N,K) row-major if transB else (K,N)
// act: 0=none, 1=relu, 2=sigmoid. Batched via grid.z with strides.
// Full in-bounds tiles are staged with global_load_async_to_lds_b128 (f32),
// then converted in LDS to f16 fragment layout; edge tiles use guarded loads.
__global__ __launch_bounds__(128) void qa_gemm_kernel(
    const float* __restrict__ A, const float* __restrict__ Bm,
    const float* __restrict__ bias, const float* __restrict__ res,
    float* __restrict__ C,
    int M, int N, int K, int lda, int ldb, int ldc,
    long sA, long sB, long sC, int transA, int transB, int act) {
  __shared__ __align__(16) float As32[64 * 36];           // [m][k] stride 36
  __shared__ __align__(16) float Bs32[64 * 36];           // transB: [n][k] s36 ; else [k][n] s68
  __shared__ __align__(16) _Float16 As16[64 * 40];        // [m][k] stride 40
  __shared__ __align__(16) _Float16 Bs16[32 * 72];        // [k][n] stride 72
  int bz = blockIdx.z;
  A += (long)bz * sA;
  Bm += (long)bz * sB;
  C += (long)bz * sC;
  const float* R = res ? (res + (long)bz * sC) : (const float*)0;
  int m0 = blockIdx.y * 64, n0 = blockIdx.x * 64;
  int tid = threadIdx.x, lane = tid & 31, wave = tid >> 5;
  int wm = (wave >> 1) * 32, wn = (wave & 1) * 32;
  bool fullM = (m0 + 64) <= M, fullN = (n0 + 64) <= N;
  unsigned aBase = qa_lds_off(As32), bBase = qa_lds_off(Bs32);
  v8f acc[2][2] = {};
  for (int k0 = 0; k0 < K; k0 += 32) {
    bool fullK = (k0 + 32) <= K;
    bool fastA = fullM && fullK && !transA && ((lda & 3) == 0);
    bool fastB = fullN && fullK && ((ldb & 3) == 0);
    if (fastA) {
      for (int i = tid; i < 512; i += 128) {
        int m = i >> 3, kq = i & 7;
        qa_async_b128(aBase + (unsigned)((m * 36 + kq * 4) * 4),
                      A + (long)(m0 + m) * lda + k0 + kq * 4);
      }
      if (k0 + 32 < K && tid < 64)
        __builtin_prefetch(A + (long)(m0 + tid) * lda + k0 + 32, 0, 1);
    } else {
      for (int i = tid; i < 2048; i += 128) {
        int m = i >> 5, kk = i & 31;
        int mm = m0 + m, k = k0 + kk;
        float v = 0.f;
        if (mm < M && k < K) v = transA ? A[(long)k * lda + mm] : A[(long)mm * lda + k];
        As32[m * 36 + kk] = v;
      }
    }
    if (fastB) {
      if (transB) {
        for (int i = tid; i < 512; i += 128) {
          int n = i >> 3, kq = i & 7;
          qa_async_b128(bBase + (unsigned)((n * 36 + kq * 4) * 4),
                        Bm + (long)(n0 + n) * ldb + k0 + kq * 4);
        }
        if (k0 + 32 < K && tid < 64)
          __builtin_prefetch(Bm + (long)(n0 + tid) * ldb + k0 + 32, 0, 1);
      } else {
        for (int i = tid; i < 512; i += 128) {
          int k = i >> 4, nq = i & 15;
          qa_async_b128(bBase + (unsigned)((k * 68 + nq * 4) * 4),
                        Bm + (long)(k0 + k) * ldb + n0 + nq * 4);
        }
      }
    } else {
      if (transB) {
        for (int i = tid; i < 2048; i += 128) {
          int n = i >> 5, kk = i & 31;
          int nn = n0 + n, k = k0 + kk;
          float v = 0.f;
          if (nn < N && k < K) v = Bm[(long)nn * ldb + k];
          Bs32[n * 36 + kk] = v;
        }
      } else {
        for (int i = tid; i < 2048; i += 128) {
          int k = i >> 6, n = i & 63;
          int nn = n0 + n, kk = k0 + k;
          float v = 0.f;
          if (nn < N && kk < K) v = Bm[(long)kk * ldb + nn];
          Bs32[k * 68 + n] = v;
        }
      }
    }
    qa_wait_async();
    __syncthreads();
    // f32 -> f16 convert into fragment layouts
    for (int i = tid; i < 2048; i += 128) {
      int m = i >> 5, kk = i & 31;
      As16[m * 40 + kk] = (_Float16)As32[m * 36 + kk];
    }
    for (int i = tid; i < 2048; i += 128) {
      int k = i >> 6, n = i & 63;
      float v = transB ? Bs32[n * 36 + k] : Bs32[k * 68 + n];
      Bs16[k * 72 + n] = (_Float16)v;
    }
    __syncthreads();
    v16h af[2], bf[2];
#pragma unroll
    for (int i = 0; i < 2; ++i) af[i] = qa_frag_a_lds(&As16[(wm + i * 16) * 40], 40, lane);
#pragma unroll
    for (int j = 0; j < 2; ++j) {
      int n = wn + j * 16 + (lane & 15);
#pragma unroll
      for (int v = 0; v < 8; ++v) {
        int kb = qa_kbase(lane, v);
        bf[j][2 * v] = Bs16[kb * 72 + n];
        bf[j][2 * v + 1] = Bs16[(kb + 1) * 72 + n];
      }
    }
#pragma unroll
    for (int i = 0; i < 2; ++i)
#pragma unroll
      for (int j = 0; j < 2; ++j)
        acc[i][j] = __builtin_amdgcn_wmma_f32_16x16x32_f16(
            false, af[i], false, bf[j], (short)0, acc[i][j], false, false);
    __syncthreads();
  }
#pragma unroll
  for (int i = 0; i < 2; ++i)
#pragma unroll
    for (int j = 0; j < 2; ++j) {
      int nn = n0 + wn + j * 16 + (lane & 15);
      if (nn >= N) continue;
      float bv = bias ? bias[nn] : 0.f;
#pragma unroll
      for (int r = 0; r < 8; ++r) {
        int mm = m0 + wm + i * 16 + r + ((lane >> 4) << 3);
        if (mm >= M) continue;
        float v = acc[i][j][r] + bv;
        if (act == 1) v = v > 0.f ? v : 0.f;
        else if (act == 2) v = 1.f / (1.f + __expf(-v));
        if (R) v += R[(long)mm * ldc + nn];
        C[(long)mm * ldc + nn] = v;
      }
    }
}

// ---------------- Fused MHA: scores(WMMA) -> masked softmax (LDS) -> A*V(WMMA) --
// Q,K,V,O: (B*L, 128) rows, head h uses cols [16h,16h+16). grid=(ceil(L/32), 8, B)
__global__ __launch_bounds__(64) void qa_attn_kernel(
    const float* __restrict__ Q, const float* __restrict__ Kt,
    const float* __restrict__ V, const float* __restrict__ maskf,
    float* __restrict__ O, int L, int Lp) {
  __shared__ float sc[32 * 416];
  __shared__ __align__(16) _Float16 Ks16[32 * 36];  // [dk(pad32)][key32] stride 36
  __shared__ __align__(16) _Float16 Vs16[32 * 20];  // [key32][dk16] stride 20
  int b = blockIdx.z, h = blockIdx.y;
  int q0 = blockIdx.x * 32;
  int lane = threadIdx.x & 31, wave = threadIdx.x >> 5;
  int h16 = h * 16;
  long base = (long)b * L * 128;
  // zero the dk=16..31 padding rows of Ks16 once
  for (int i = threadIdx.x; i < 16 * 36; i += 64) Ks16[16 * 36 + i] = (_Float16)0.f;
  // Q fragment (dk=16, zero-padded to K=32)
  v16h aq;
  {
    int m = q0 + wave * 16 + (lane & 15);
    const float* qrow = Q + base + (long)m * 128 + h16;
    int ok = (m < L);
#pragma unroll
    for (int v = 0; v < 8; ++v) {
      int kb = qa_kbase(lane, v);
      float x0 = (ok && kb < 16) ? qrow[kb] : 0.f;
      float x1 = (ok && kb < 16) ? qrow[kb + 1] : 0.f;
      aq[2 * v] = (_Float16)x0;
      aq[2 * v + 1] = (_Float16)x1;
    }
  }
  __syncthreads();
  // scores in 32-key chunks: stage K^T tile (coalesced), 2 WMMAs per chunk
  for (int kc = 0; kc < Lp; kc += 32) {
    for (int i = threadIdx.x; i < 512; i += 64) {
      int dk = i & 15, kr = i >> 4, key = kc + kr;
      float v = (key < L) ? Kt[base + (long)key * 128 + h16 + dk] : 0.f;
      Ks16[dk * 36 + kr] = (_Float16)v;
    }
    __syncthreads();
#pragma unroll
    for (int sub = 0; sub < 2; ++sub) {
      v16h bf;
#pragma unroll
      for (int v = 0; v < 8; ++v) {
        int kb = qa_kbase(lane, v);
        bf[2 * v] = Ks16[kb * 36 + sub * 16 + (lane & 15)];
        bf[2 * v + 1] = Ks16[(kb + 1) * 36 + sub * 16 + (lane & 15)];
      }
      v8f c = {};
      c = __builtin_amdgcn_wmma_f32_16x16x32_f16(false, aq, false, bf, (short)0, c, false, false);
      int key0 = kc + sub * 16;
#pragma unroll
      for (int r = 0; r < 8; ++r) {
        int ml = wave * 16 + r + ((lane >> 4) << 3);
        int key = key0 + (lane & 15);
        float mk = (key < L) ? maskf[b * L + key] : 0.f;
        float s = c[r] * 0.25f;  // dk^-0.5, dk=16
        s = s * mk + NEGBIG * (1.f - mk);
        sc[ml * 416 + key] = s;
      }
    }
    __syncthreads();
  }
  // per-row softmax: 2 lanes per row, wave-private rows -> no barrier needed
  {
    int r = lane & 15, half = lane >> 4;
    float* row = &sc[(wave * 16 + r) * 416];
    int j0 = half * (Lp >> 1), j1 = j0 + (Lp >> 1);
    float mx = NEGBIG;
    for (int j = j0; j < j1; ++j) mx = fmaxf(mx, row[j]);
    mx = fmaxf(mx, __shfl_xor(mx, 16, 32));
    float sum = 0.f;
    for (int j = j0; j < j1; ++j) sum += __expf(row[j] - mx);
    sum += __shfl_xor(sum, 16, 32);
    float inv = 1.f / sum;
    for (int j = j0; j < j1; ++j) row[j] = __expf(row[j] - mx) * inv;
  }
  // O = P * V with staged V tiles
  v8f acc = {};
  for (int kc = 0; kc < Lp; kc += 32) {
    __syncthreads();
    for (int i = threadIdx.x; i < 512; i += 64) {
      int dk = i & 15, kr = i >> 4, key = kc + kr;
      float v = (key < L) ? V[base + (long)key * 128 + h16 + dk] : 0.f;
      Vs16[kr * 20 + dk] = (_Float16)v;
    }
    __syncthreads();
    v16h ap;
    {
      const float* row = &sc[(wave * 16 + (lane & 15)) * 416 + kc];
#pragma unroll
      for (int v = 0; v < 8; ++v) {
        int kb = qa_kbase(lane, v);
        ap[2 * v] = (_Float16)row[kb];
        ap[2 * v + 1] = (_Float16)row[kb + 1];
      }
    }
    v16h bv;
#pragma unroll
    for (int v = 0; v < 8; ++v) {
      int kb = qa_kbase(lane, v);
      bv[2 * v] = Vs16[kb * 20 + (lane & 15)];
      bv[2 * v + 1] = Vs16[(kb + 1) * 20 + (lane & 15)];
    }
    acc = __builtin_amdgcn_wmma_f32_16x16x32_f16(false, ap, false, bv, (short)0, acc, false, false);
  }
#pragma unroll
  for (int r = 0; r < 8; ++r) {
    int mm = q0 + wave * 16 + r + ((lane >> 4) << 3);
    if (mm < L) O[base + (long)mm * 128 + h16 + (lane & 15)] = acc[r];
  }
}

// ---------------- elementwise / small kernels ----------------
__global__ void qa_mask_kernel(const int* __restrict__ idx, float* __restrict__ m, long n) {
  long i = blockIdx.x * (long)blockDim.x + threadIdx.x;
  if (i < n) m[i] = (idx[i] != 0) ? 1.f : 0.f;
}

__global__ void qa_gather_kernel(const int* __restrict__ ids, const float* __restrict__ tab,
                                 float* __restrict__ out, long rows, int dim, int ostride, int ooff) {
  long i = blockIdx.x * (long)blockDim.x + threadIdx.x;
  if (i >= rows * (long)dim) return;
  int c = (int)(i % dim);
  long r = i / dim;
  out[r * (long)ostride + ooff + c] = tab[(long)ids[r] * dim + c];
}

__global__ void qa_dws2_kernel(const float* __restrict__ X, const float* __restrict__ W,
                               const float* __restrict__ bias, float* __restrict__ Y,
                               int Bn, int L, int Tn, int C) {
  long i = blockIdx.x * (long)blockDim.x + threadIdx.x;
  long total = (long)Bn * L * Tn * C;
  if (i >= total) return;
  int c = (int)(i % C);
  long r = i / C;
  int t = (int)(r % Tn); r /= Tn;
  int l = (int)(r % L);
  int b = (int)(r / L);
  float s = bias[c];
#pragma unroll
  for (int dl = 0; dl < 5; ++dl)
#pragma unroll
    for (int dt = 0; dt < 5; ++dt) {
      int ls = l + dl - 2, ts = t + dt - 2;
      if (ls >= 0 && ls < L && ts >= 0 && ts < Tn)
        s += W[c * 25 + dl * 5 + dt] * X[(((long)b * L + ls) * Tn + ts) * C + c];
    }
  Y[i] = s;
}

__global__ void qa_maxpool_kernel(const float* __restrict__ X, float* __restrict__ out,
                                  long rows, int Tn, int C, int ostride) {
  long i = blockIdx.x * (long)blockDim.x + threadIdx.x;
  if (i >= rows * (long)C) return;
  int c = (int)(i % C);
  long r = i / C;
  float m = -3.4e38f;
  for (int t = 0; t < Tn; ++t) m = fmaxf(m, X[((long)r * Tn + t) * C + c]);
  out[r * (long)ostride + c] = m;
}

__global__ void qa_dws1_kernel(const float* __restrict__ X, const float* __restrict__ W,
                               const float* __restrict__ bias, float* __restrict__ Y,
                               int Bn, int L, int C, int k) {
  long i = blockIdx.x * (long)blockDim.x + threadIdx.x;
  long total = (long)Bn * L * C;
  if (i >= total) return;
  int c = (int)(i % C);
  long bl = i / C;
  int l = (int)(bl % L);
  int b = (int)(bl / L);
  int half = k >> 1;
  float s = bias[c];
  for (int t = 0; t < k; ++t) {
    int ls = l + t - half;
    if (ls >= 0 && ls < L) s += W[c * k + t] * X[((long)b * L + ls) * C + c];
  }
  Y[i] = s;
}

__global__ void qa_posenc_kernel(float* __restrict__ X, long rows, int L) {
  long i = blockIdx.x * (long)blockDim.x + threadIdx.x;
  if (i >= rows * 128L) return;
  int c = (int)(i % 128);
  long r = i / 128;
  int l = (int)(r % L);
  float e = -(float)(l & ~1) / 128.f;
  X[i] += sinf((float)c * __powf(10000.f, e));
}

__global__ void qa_layernorm_kernel(const float* __restrict__ X, const float* __restrict__ g,
                                    const float* __restrict__ bt, float* __restrict__ Y, long rows) {
  long row = blockIdx.x * 4L + (threadIdx.x >> 5);
  int lane = threadIdx.x & 31;
  if (row >= rows) return;
  const float* x = X + row * 128;
  float v0 = x[lane], v1 = x[lane + 32], v2 = x[lane + 64], v3 = x[lane + 96];
  float s = v0 + v1 + v2 + v3;
  for (int o = 16; o; o >>= 1) s += __shfl_xor(s, o, 32);
  float mu = s * (1.f / 128.f);
  float d0 = v0 - mu, d1 = v1 - mu, d2 = v2 - mu, d3 = v3 - mu;
  float q = d0 * d0 + d1 * d1 + d2 * d2 + d3 * d3;
  for (int o = 16; o; o >>= 1) q += __shfl_xor(q, o, 32);
  float inv = rsqrtf(q * (1.f / 128.f) + 1e-5f);
  float* y = Y + row * 128;
  y[lane] = d0 * inv * g[lane] + bt[lane];
  y[lane + 32] = d1 * inv * g[lane + 32] + bt[lane + 32];
  y[lane + 64] = d2 * inv * g[lane + 64] + bt[lane + 64];
  y[lane + 96] = d3 * inv * g[lane + 96] + bt[lane + 96];
}

__global__ void qa_highway_kernel(float* __restrict__ x, const float* __restrict__ g,
                                  const float* __restrict__ t, long n) {
  long i = blockIdx.x * (long)blockDim.x + threadIdx.x;
  if (i < n) x[i] = g[i] * t[i] + (1.f - g[i]) * x[i];
}

__global__ void qa_rowdot_kernel(const float* __restrict__ X, const float* __restrict__ w,
                                 float* __restrict__ out, long rows) {
  long row = blockIdx.x * 4L + (threadIdx.x >> 5);
  int lane = threadIdx.x & 31;
  if (row >= rows) return;
  const float* x = X + row * 128;
  float s = x[lane] * w[lane] + x[lane + 32] * w[lane + 32] +
            x[lane + 64] * w[lane + 64] + x[lane + 96] * w[lane + 96];
  for (int o = 16; o; o >>= 1) s += __shfl_xor(s, o, 32);
  if (lane == 0) out[row] = s;
}

__global__ void qa_scale_kernel(const float* __restrict__ X, const float* __restrict__ w,
                                float* __restrict__ Y, long n) {
  long i = blockIdx.x * (long)blockDim.x + threadIdx.x;
  if (i < n) Y[i] = X[i] * w[i & 127];
}

__global__ void qa_softmax_j_kernel(const float* __restrict__ S, const float* __restrict__ cw,
                                    const float* __restrict__ qw, const float* __restrict__ qmask,
                                    float* __restrict__ D, int CL, int QL) {
  int i = blockIdx.x, b = blockIdx.y, lane = threadIdx.x;
  const float* srow = S + ((long)b * CL + i) * QL;
  float ci = cw[b * CL + i];
  float mx = NEGBIG;
  for (int j = lane; j < QL; j += 32) {
    float m = qmask[b * QL + j];
    float v = (srow[j] + ci + qw[b * QL + j]) * m + NEGBIG * (1.f - m);
    mx = fmaxf(mx, v);
  }
  for (int o = 16; o; o >>= 1) mx = fmaxf(mx, __shfl_xor(mx, o, 32));
  float sum = 0.f;
  for (int j = lane; j < QL; j += 32) {
    float m = qmask[b * QL + j];
    float v = (srow[j] + ci + qw[b * QL + j]) * m + NEGBIG * (1.f - m);
    sum += __expf(v - mx);
  }
  for (int o = 16; o; o >>= 1) sum += __shfl_xor(sum, o, 32);
  float inv = 1.f / sum;
  float* drow = D + ((long)b * CL + i) * QL;
  for (int j = lane; j < QL; j += 32) {
    float m = qmask[b * QL + j];
    float v = (srow[j] + ci + qw[b * QL + j]) * m + NEGBIG * (1.f - m);
    drow[j] = __expf(v - mx) * inv;
  }
}

__global__ void qa_softmax_i_kernel(const float* __restrict__ S, const float* __restrict__ cw,
                                    const float* __restrict__ qw, const float* __restrict__ cmask,
                                    float* __restrict__ D, int CL, int QL) {
  int j = blockIdx.x, b = blockIdx.y, lane = threadIdx.x;
  float qj = qw[b * QL + j];
  float mx = NEGBIG;
  for (int i = lane; i < CL; i += 32) {
    float m = cmask[b * CL + i];
    float v = (S[((long)b * CL + i) * QL + j] + cw[b * CL + i] + qj) * m + NEGBIG * (1.f - m);
    mx = fmaxf(mx, v);
  }
  for (int o = 16; o; o >>= 1) mx = fmaxf(mx, __shfl_xor(mx, o, 32));
  float sum = 0.f;
  for (int i = lane; i < CL; i += 32) {
    float m = cmask[b * CL + i];
    float v = (S[((long)b * CL + i) * QL + j] + cw[b * CL + i] + qj) * m + NEGBIG * (1.f - m);
    sum += __expf(v - mx);
  }
  for (int o = 16; o; o >>= 1) sum += __shfl_xor(sum, o, 32);
  float inv = 1.f / sum;
  for (int i = lane; i < CL; i += 32) {
    float m = cmask[b * CL + i];
    float v = (S[((long)b * CL + i) * QL + j] + cw[b * CL + i] + qj) * m + NEGBIG * (1.f - m);
    D[((long)b * CL + i) * QL + j] = __expf(v - mx) * inv;
  }
}

__global__ void qa_concat4_kernel(const float* __restrict__ Ct, const float* __restrict__ Am,
                                  const float* __restrict__ Bm, float* __restrict__ out, long rows) {
  long i = blockIdx.x * (long)blockDim.x + threadIdx.x;
  if (i >= rows * 512L) return;
  int c = (int)(i % 512);
  long r = i / 512;
  const float* ct = Ct + r * 128;
  float v;
  if (c < 128) v = ct[c];
  else if (c < 256) v = Am[r * 128 + c - 128];
  else if (c < 384) v = ct[c - 256] * Am[r * 128 + c - 256];
  else v = ct[c - 384] * Bm[r * 128 + c - 384];
  out[i] = v;
}

__global__ __launch_bounds__(256) void qa_decoder_kernel(
    const float* __restrict__ M0, const float* __restrict__ Mx, const float* __restrict__ W,
    const float* __restrict__ maskf, float* __restrict__ out, int L) {
  __shared__ float ybuf[400];
  __shared__ float red[256];
  int b = blockIdx.x, t = threadIdx.x;
  for (int l = t; l < L; l += 256) {
    long row = (long)(b * L + l) * 128;
    float s = 0.f;
    for (int c = 0; c < 128; ++c) s += W[c] * M0[row + c] + W[128 + c] * Mx[row + c];
    float mk = maskf[b * L + l];
    ybuf[l] = s * mk + NEGBIG * (1.f - mk);
  }
  __syncthreads();
  float mx = NEGBIG;
  for (int l = t; l < L; l += 256) mx = fmaxf(mx, ybuf[l]);
  red[t] = mx; __syncthreads();
  for (int s = 128; s > 0; s >>= 1) { if (t < s) red[t] = fmaxf(red[t], red[t + s]); __syncthreads(); }
  mx = red[0]; __syncthreads();
  float sum = 0.f;
  for (int l = t; l < L; l += 256) sum += __expf(ybuf[l] - mx);
  red[t] = sum; __syncthreads();
  for (int s = 128; s > 0; s >>= 1) { if (t < s) red[t] += red[t + s]; __syncthreads(); }
  float lse = __logf(red[0]) + mx;
  for (int l = t; l < L; l += 256) out[(long)b * L + l] = ybuf[l] - lse;
}

// ---------------- host orchestration ----------------
struct QaLin { const float* w; const float* b; };
struct QaDws { const float* dww; const float* dwb; const float* pww; const float* pwb; };
struct QaLn  { const float* g; const float* b; };
struct QaEnc { QaDws conv[4]; QaLn ln[4]; QaLin q, k, v, f; QaLn ln2, ln3; QaLin fc; };

extern "C" void kernel_launch(void* const* d_in, const int* in_sizes, int n_in,
                              void* d_out, int out_size, void* d_ws, size_t ws_size,
                              hipStream_t stream) {
  (void)in_sizes; (void)out_size; (void)ws_size;
  const int B = 32, CL = 400, QL = 50, CHL = 16, WD = 300, MD = 128, HC = 364, NH = 8;
  if (n_in < 147) return;
  int p = 0;
  auto F = [&](int i) { return (const float*)d_in[i]; };
  const float* word_vec = F(p++);
  const float* char_vec = F(p++);
  QaDws conv2 = {F(p), F(p + 1), F(p + 2), F(p + 3)}; p += 4;
  QaLin gates[2], trans[2];
  for (int i = 0; i < 2; ++i) { gates[i] = {F(p), F(p + 1)}; p += 2; }
  for (int i = 0; i < 2; ++i) { trans[i] = {F(p), F(p + 1)}; p += 2; }
  QaDws cconv = {F(p), F(p + 1), F(p + 2), F(p + 3)}; p += 4;
  QaDws qconv = {F(p), F(p + 1), F(p + 2), F(p + 3)}; p += 4;
  QaEnc encC, encQ, encM;
  auto parseEnc = [&](QaEnc& e) {
    for (int i = 0; i < 4; ++i) { e.conv[i] = {F(p), F(p + 1), F(p + 2), F(p + 3)}; p += 4; }
    for (int i = 0; i < 4; ++i) { e.ln[i] = {F(p), F(p + 1)}; p += 2; }
    e.q = {F(p), F(p + 1)}; p += 2;
    e.k = {F(p), F(p + 1)}; p += 2;
    e.v = {F(p), F(p + 1)}; p += 2;
    e.f = {F(p), F(p + 1)}; p += 2;
    e.ln2 = {F(p), F(p + 1)}; p += 2;
    e.ln3 = {F(p), F(p + 1)}; p += 2;
    e.fc = {F(p), F(p + 1)}; p += 2;
  };
  parseEnc(encC);
  parseEnc(encQ);
  const float* W0 = F(p++);
  QaDws resizer = {F(p), F(p + 1), F(p + 2), F(p + 3)}; p += 4;
  parseEnc(encM);
  const float* W1 = F(p++);
  const float* W2 = F(p++);
  const int* cw_idx = (const int*)d_in[p++];
  const int* cc_idx = (const int*)d_in[p++];
  const int* qw_idx = (const int*)d_in[p++];
  const int* qc_idx = (const int*)d_in[p++];

  // workspace bump allocator
  size_t off = 0;
  auto alloc = [&](long n) -> float* {
    float* q = (float*)((char*)d_ws + off);
    off += ((size_t)n * 4 + 255) & ~(size_t)255;
    return q;
  };
  float* cmaskf = alloc((long)B * CL);
  float* qmaskf = alloc((long)B * QL);
  float* ch1 = alloc((long)B * CL * CHL * 64);
  float* ch2 = alloc((long)B * CL * CHL * 64);
  float* xtc = alloc((long)B * CL * HC);
  float* xtq = alloc((long)B * QL * HC);
  float* hg = alloc((long)B * CL * HC);
  float* ht = alloc((long)B * CL * HC);
  float* t512 = alloc((long)B * CL * 512);  // depthwise temp (364 & 512 ch)
  float* Xb = alloc((long)B * CL * MD);
  float* X2b = alloc((long)B * CL * MD);
  float* Yb = alloc((long)B * CL * MD);
  float* Tb = alloc((long)B * CL * MD);
  float* Qb2 = alloc((long)B * CL * MD);
  float* Kb2 = alloc((long)B * CL * MD);
  float* Vb2 = alloc((long)B * CL * MD);
  float* Ob2 = alloc((long)B * CL * MD);
  float* Cbuf = alloc((long)B * CL * MD);
  float* Qbuf = alloc((long)B * QL * MD);
  float* cwv = alloc((long)B * CL);
  float* qwv = alloc((long)B * QL);
  float* Cm = alloc((long)B * CL * MD);
  float* Sb = alloc((long)B * CL * QL);
  float* Sbar = alloc((long)B * CL * QL);
  float* Sbbar = alloc((long)B * CL * QL);
  float* Am = alloc((long)B * CL * MD);
  float* Tm = alloc((long)B * QL * MD);
  float* Bm2 = alloc((long)B * CL * MD);
  float* Xcat = alloc((long)B * CL * 512);
  float* M0 = alloc((long)B * CL * MD);
  float* M1 = alloc((long)B * CL * MD);
  float* M2 = alloc((long)B * CL * MD);

  auto ew = [](long n) { return dim3((unsigned)((n + 255) / 256)); };
  auto gemm = [&](const float* A, const float* Bm, const float* bias, const float* res, float* C,
                  int M, int N, int K, int lda, int ldb, int ldc, int batch,
                  long sA, long sB, long sC, int tA, int tB, int act) {
    dim3 g((N + 63) / 64, (M + 63) / 64, batch);
    qa_gemm_kernel<<<g, dim3(128), 0, stream>>>(A, Bm, bias, res, C, M, N, K, lda, ldb, ldc,
                                                sA, sB, sC, tA, tB, act);
  };

  qa_mask_kernel<<<ew((long)B * CL), 256, 0, stream>>>(cw_idx, cmaskf, (long)B * CL);
  qa_mask_kernel<<<ew((long)B * QL), 256, 0, stream>>>(qw_idx, qmaskf, (long)B * QL);

  auto embed = [&](const int* widx, const int* cidx, int L, float* xt) {
    long rows = (long)B * L, crows = rows * CHL;
    qa_gather_kernel<<<ew(crows * 64), 256, 0, stream>>>(cidx, char_vec, ch1, crows, 64, 64, 0);
    qa_dws2_kernel<<<ew(crows * 64), 256, 0, stream>>>(ch1, conv2.dww, conv2.dwb, ch2, B, L, CHL, 64);
    gemm(ch2, conv2.pww, conv2.pwb, nullptr, ch1, (int)crows, 64, 64, 64, 64, 64, 1, 0, 0, 0, 0, 1, 1);
    qa_maxpool_kernel<<<ew(rows * 64), 256, 0, stream>>>(ch1, xt, rows, CHL, 64, HC);
    qa_gather_kernel<<<ew(rows * WD), 256, 0, stream>>>(widx, word_vec, xt, rows, WD, HC, 64);
    for (int i = 0; i < 2; ++i) {
      gemm(xt, gates[i].w, gates[i].b, nullptr, hg, (int)rows, HC, HC, HC, HC, HC, 1, 0, 0, 0, 0, 1, 2);
      gemm(xt, trans[i].w, trans[i].b, nullptr, ht, (int)rows, HC, HC, HC, HC, HC, 1, 0, 0, 0, 0, 1, 1);
      qa_highway_kernel<<<ew(rows * HC), 256, 0, stream>>>(xt, hg, ht, rows * (long)HC);
    }
  };

  auto encoder = [&](const QaEnc& e, const float* maskf, int L) {
    long rows = (long)B * L;
    int Lp = (L + 31) & ~31;
    qa_posenc_kernel<<<ew(rows * 128), 256, 0, stream>>>(Xb, rows, L);
    float* x = Xb;
    float* x2 = X2b;
    for (int i = 0; i < 4; ++i) {
      qa_layernorm_kernel<<<(unsigned)((rows + 3) / 4), 128, 0, stream>>>(x, e.ln[i].g, e.ln[i].b, Yb, rows);
      qa_dws1_kernel<<<ew(rows * 128), 256, 0, stream>>>(Yb, e.conv[i].dww, e.conv[i].dwb, Tb, B, L, 128, 7);
      gemm(Tb, e.conv[i].pww, e.conv[i].pwb, x, x2, (int)rows, 128, 128, 128, 128, 128, 1, 0, 0, 0, 0, 1, 1);
      float* t = x; x = x2; x2 = t;
    }
    qa_layernorm_kernel<<<(unsigned)((rows + 3) / 4), 128, 0, stream>>>(x, e.ln2.g, e.ln2.b, Yb, rows);
    gemm(Yb, e.q.w, e.q.b, nullptr, Qb2, (int)rows, 128, 128, 128, 128, 128, 1, 0, 0, 0, 0, 1, 0);
    gemm(Yb, e.k.w, e.k.b, nullptr, Kb2, (int)rows, 128, 128, 128, 128, 128, 1, 0, 0, 0, 0, 1, 0);
    gemm(Yb, e.v.w, e.v.b, nullptr, Vb2, (int)rows, 128, 128, 128, 128, 128, 1, 0, 0, 0, 0, 1, 0);
    qa_attn_kernel<<<dim3((L + 31) / 32, NH, B), 64, 0, stream>>>(Qb2, Kb2, Vb2, maskf, Ob2, L, Lp);
    gemm(Ob2, e.f.w, e.f.b, x, x2, (int)rows, 128, 128, 128, 128, 128, 1, 0, 0, 0, 0, 1, 0);
    { float* t = x; x = x2; x2 = t; }
    qa_layernorm_kernel<<<(unsigned)((rows + 3) / 4), 128, 0, stream>>>(x, e.ln3.g, e.ln3.b, Yb, rows);
    gemm(Yb, e.fc.w, e.fc.b, x, x2, (int)rows, 128, 128, 128, 128, 128, 1, 0, 0, 0, 0, 1, 1);
    // after 6 swaps result is back in Xb
  };

  // ---- context branch ----
  embed(cw_idx, cc_idx, CL, xtc);
  qa_dws1_kernel<<<ew((long)B * CL * HC), 256, 0, stream>>>(xtc, cconv.dww, cconv.dwb, t512, B, CL, HC, 5);
  gemm(t512, cconv.pww, cconv.pwb, nullptr, Xb, B * CL, 128, HC, HC, HC, 128, 1, 0, 0, 0, 0, 1, 0);
  encoder(encC, cmaskf, CL);
  hipMemcpyAsync(Cbuf, Xb, (long)B * CL * 128 * 4, hipMemcpyDeviceToDevice, stream);
  // ---- question branch ----
  embed(qw_idx, qc_idx, QL, xtq);
  qa_dws1_kernel<<<ew((long)B * QL * HC), 256, 0, stream>>>(xtq, qconv.dww, qconv.dwb, t512, B, QL, HC, 5);
  gemm(t512, qconv.pww, qconv.pwb, nullptr, Xb, B * QL, 128, HC, HC, HC, 128, 1, 0, 0, 0, 0, 1, 0);
  encoder(encQ, qmaskf, QL);
  hipMemcpyAsync(Qbuf, Xb, (long)B * QL * 128 * 4, hipMemcpyDeviceToDevice, stream);
  // ---- co-attention ----
  qa_rowdot_kernel<<<(unsigned)((B * CL + 3) / 4), 128, 0, stream>>>(Cbuf, W0, cwv, (long)B * CL);
  qa_rowdot_kernel<<<(unsigned)((B * QL + 3) / 4), 128, 0, stream>>>(Qbuf, W0 + 128, qwv, (long)B * QL);
  qa_scale_kernel<<<ew((long)B * CL * 128), 256, 0, stream>>>(Cbuf, W0 + 256, Cm, (long)B * CL * 128);
  gemm(Cm, Qbuf, nullptr, nullptr, Sb, CL, QL, 128, 128, 128, QL, B,
       (long)CL * 128, (long)QL * 128, (long)CL * QL, 0, 1, 0);
  qa_softmax_j_kernel<<<dim3(CL, B), 32, 0, stream>>>(Sb, cwv, qwv, qmaskf, Sbar, CL, QL);
  qa_softmax_i_kernel<<<dim3(QL, B), 32, 0, stream>>>(Sb, cwv, qwv, cmaskf, Sbbar, CL, QL);
  gemm(Sbar, Qbuf, nullptr, nullptr, Am, CL, 128, QL, QL, 128, 128, B,
       (long)CL * QL, (long)QL * 128, (long)CL * 128, 0, 0, 0);
  gemm(Sbbar, Cbuf, nullptr, nullptr, Tm, QL, 128, CL, QL, 128, 128, B,
       (long)CL * QL, (long)CL * 128, (long)QL * 128, 1, 0, 0);  // Tm = S_bbar^T * Ct
  gemm(Sbar, Tm, nullptr, nullptr, Bm2, CL, 128, QL, QL, 128, 128, B,
       (long)CL * QL, (long)QL * 128, (long)CL * 128, 0, 0, 0);  // Bm = S_bar * Tm
  qa_concat4_kernel<<<ew((long)B * CL * 512), 256, 0, stream>>>(Cbuf, Am, Bm2, Xcat, (long)B * CL);
  qa_dws1_kernel<<<ew((long)B * CL * 512), 256, 0, stream>>>(Xcat, resizer.dww, resizer.dwb, t512, B, CL, 512, 5);
  gemm(t512, resizer.pww, resizer.pwb, nullptr, Xb, B * CL, 128, 512, 512, 512, 128, 1, 0, 0, 0, 0, 1, 0);
  // ---- model encoders ----
  encoder(encM, cmaskf, CL);
  hipMemcpyAsync(M0, Xb, (long)B * CL * 128 * 4, hipMemcpyDeviceToDevice, stream);
  encoder(encM, cmaskf, CL);
  hipMemcpyAsync(M1, Xb, (long)B * CL * 128 * 4, hipMemcpyDeviceToDevice, stream);
  encoder(encM, cmaskf, CL);
  hipMemcpyAsync(M2, Xb, (long)B * CL * 128 * 4, hipMemcpyDeviceToDevice, stream);
  // ---- decoder ----
  float* out = (float*)d_out;
  qa_decoder_kernel<<<dim3(B), 256, 0, stream>>>(M0, M1, W1, cmaskf, out, CL);
  qa_decoder_kernel<<<dim3(B), 256, 0, stream>>>(M0, M2, W2, cmaskf, out + (long)B * CL, CL);
}
